// TileRefineRAFTLike_49684181680667
// MI455X (gfx1250) — matrix operand order repaired and checked by reading, not compiled, not run on hardware
//
#include <hip/hip_runtime.h>
#include <stdint.h>

typedef __attribute__((ext_vector_type(16))) __bf16 v16bf;
typedef __attribute__((ext_vector_type(8)))  float  v8f;

constexpr int Bn = 4, Hh = 128, Ww = 256;
constexpr int HW = Hh * Ww;
constexpr int GRUIN = 236;

__device__ __forceinline__ unsigned short f2bf(float f) {
  union { float f; uint32_t u; } x; x.f = f;
  uint32_t r = (x.u + 0x7FFFu + ((x.u >> 16) & 1u)) >> 16;
  return (unsigned short)r;
}
__device__ __forceinline__ float bf2f(unsigned short h) {
  union { uint32_t u; float f; } x; x.u = ((uint32_t)h) << 16;
  return x.f;
}
__device__ __forceinline__ float sigm(float x) { return 1.f / (1.f + __expf(-x)); }

// ---------------------------------------------------------------------------
// Stage 1: warp-sample fR, cost volume, assemble bf16 GRU input.
// bFeat: [B][64][HW]  (feat)
// bCtx : [B][192][HW] (fL 0-63 | fR_w 64-127 | d,sx,sy,conf 128-131 |
//                      cost 132-171 | zero-pad 172-191)
// ---------------------------------------------------------------------------
__global__ __launch_bounds__(256) void buildCtx(
    const float* __restrict__ dI, const float* __restrict__ sxI,
    const float* __restrict__ syI, const float* __restrict__ confI,
    const float* __restrict__ feat, const float* __restrict__ fL,
    const float* __restrict__ fR,
    unsigned short* __restrict__ bFeat, unsigned short* __restrict__ bCtx) {
  int gid = blockIdx.x * 256 + threadIdx.x;
  if (gid >= Bn * HW) return;
  int b = gid / HW, p = gid % HW;
  int y = p / Ww, x = p % Ww;
  float d0 = dI[gid];

  for (int c = 0; c < 64; ++c) {
    bFeat[((size_t)b * 64 + c) * HW + p]  = f2bf(feat[((size_t)b * 64 + c) * HW + p]);
    bCtx [((size_t)b * 192 + c) * HW + p] = f2bf(fL[((size_t)b * 64 + c) * HW + p]);
  }
  bCtx[((size_t)b * 192 + 128) * HW + p] = f2bf(d0);
  bCtx[((size_t)b * 192 + 129) * HW + p] = f2bf(sxI[gid]);
  bCtx[((size_t)b * 192 + 130) * HW + p] = f2bf(syI[gid]);
  bCtx[((size_t)b * 192 + 131) * HW + p] = f2bf(confI[gid]);

  for (int dd = 0; dd < 5; ++dd) {
    float xs  = (float)x - (d0 + (float)(dd - 2));
    float xc  = fminf(fmaxf(xs, 0.f), (float)(Ww - 1));
    float x0f = floorf(xc);
    float w1  = xc - x0f;
    int   i0  = (int)x0f;
    int   i1  = min(i0 + 1, Ww - 1);
    float cg[8] = {0.f, 0.f, 0.f, 0.f, 0.f, 0.f, 0.f, 0.f};
#pragma unroll
    for (int c = 0; c < 64; ++c) {   // unrolled -> cg index is constant
      const float* row = fR + ((size_t)(b * 64 + c) * Hh + y) * Ww;
      float v = row[i0] * (1.f - w1) + row[i1] * w1;
      cg[c >> 3] += fL[((size_t)b * 64 + c) * HW + p] * v;
      if (dd == 2)  // offset 0 == fR_w
        bCtx[((size_t)b * 192 + 64 + c) * HW + p] = f2bf(v);
    }
    for (int g = 0; g < 8; ++g)
      bCtx[((size_t)b * 192 + 132 + g * 5 + dd) * HW + p] = f2bf(cg[g] * 0.125f);
  }
  for (int c = 172; c < 192; ++c)
    bCtx[((size_t)b * 192 + c) * HW + p] = 0;
}

// ---------------------------------------------------------------------------
// Weight pre-pack: f32 [Co][CinReal][3][3] -> bf16 A-fragments matching the
// 16x32 A tile in contiguous k order (k = 16*(lane>>4) + j), as the B-frags use, contiguous 16 elems per (tile,lane).
// K ordering: kt = chunk*9 + (ky*3+kx); within-tile k = channel-in-chunk.
// ---------------------------------------------------------------------------
__global__ __launch_bounds__(256) void packW(
    const float* __restrict__ w0, const float* __restrict__ w1,
    unsigned short* __restrict__ out,
    int KT, int COT, int CinReal, int CoReal, int CoSplit) {
  int t = blockIdx.x * 256 + threadIdx.x;
  int total = KT * COT * 32;
  if (t >= total) return;
  int lane = t & 31, tile = t >> 5;
  int cot = tile % COT, kt = tile / COT;
  int chunk = kt / 9, kidx = kt % 9;
  int half = lane >> 4, m = lane & 15;
  int co = cot * 16 + m;
  unsigned short* o = out + (size_t)t * 16;
  for (int j = 0; j < 16; ++j) {
    int v = j >> 1, pos = j & 1;
    // lanes 0-15: K {0..15}; lanes 16-31: K {16..31}
    int klo = half * 16 + v * 2 + pos;
    int ci = chunk * 32 + klo;
    float val = 0.f;
    if (ci < CinReal && co < CoReal) {
      const float* w = (co < CoSplit) ? w0 : w1;
      int coo = (co < CoSplit) ? co : co - CoSplit;
      val = w[((size_t)coo * CinReal + ci) * 9 + kidx];
    }
    o[j] = f2bf(val);
  }
}

// ---------------------------------------------------------------------------
// Implicit-GEMM 3x3 conv via v_wmma_f32_16x16x32_bf16.
// Block: 256 threads (8 waves) -> 64 output pixels of one row, all Co.
// Per 32-channel chunk: stage raw 3x66x32 tile once into LDS
// (channel-contiguous per pixel, pixel stride padded to 40 shorts), then all
// 9 taps read B-fragments from the same tile at immediate offsets.
// wave w: pixel tile = w&3 (16 px); co tiles = [(w>>2)*NACC, +NACC)
// MODE 0: z/r gates (Co=128): z=sigmoid -> outF(f32), r*feat -> outB(bf16)
// MODE 1: q gate   (Co=64):  h=(1-z)*feat+z*tanh(q) -> outF(f32) + outB(bf16)
// MODE 2: raw      (Co padded to 64, stores only co<48) -> outF(f32)
// ---------------------------------------------------------------------------
template <int MODE, int COT, int CHUNKS>
__global__ __launch_bounds__(256) void convWMMA(
    const unsigned short* __restrict__ in0, int nc0,
    const unsigned short* __restrict__ in1, int nc1,
    const unsigned short* __restrict__ wpack,
    const float* __restrict__ bias0, const float* __restrict__ bias1,
    const float* __restrict__ auxFeat, const float* __restrict__ auxZ,
    float* __restrict__ outF, unsigned short* __restrict__ outB) {
  static_assert((COT & 1) == 0, "COT must be even");
  constexpr int NACC = COT / 2;
  constexpr int PXS = 40;          // padded channel dim (shorts) per pixel
  constexpr int ROWS = 66 * PXS;   // shorts per tile row
  const int tid  = threadIdx.x;
  const int lane = tid & 31;
  const int wv   = tid >> 5;
  const int pxt  = wv & 3;
  const int cotB = (wv >> 2) * NACC;  // first co-tile of this wave

  const int b = blockIdx.z, y = blockIdx.y, x0 = blockIdx.x * 64;

  v8f acc[NACC];
  const v8f vzero = {0.f, 0.f, 0.f, 0.f, 0.f, 0.f, 0.f, 0.f};
#pragma unroll
  for (int i = 0; i < NACC; ++i) acc[i] = vzero;

  __shared__ __align__(16) unsigned short sT[3][66][PXS];

  // B-frag base: tile col = (output px within tile) ; taps add (ky*ROWS+kx*PXS)
  const unsigned short* bBase = &sT[0][pxt * 16 + (lane & 15)][(lane >> 4) * 16];

#pragma unroll 1
  for (int chunk = 0; chunk < CHUNKS; ++chunk) {
    const unsigned short* src;
    int cb, nc;
    if (chunk < 2) { src = in0; cb = chunk * 32; nc = nc0; }
    else           { src = in1; cb = chunk * 32 - 64; nc = nc1; }

    __syncthreads();
    // interior fill: 64 cols x 32 ch x 3 rows; x always in-range, b32 loads
#pragma unroll
    for (int row = 0; row < 3; ++row) {
      const int yy = y + row - 1;
      const bool rowOK = (yy >= 0) && (yy < Hh);
#pragma unroll
      for (int i = 0; i < 4; ++i) {
        int idx = i * 256 + tid;
        int ci  = idx >> 5;
        int px2 = (idx & 31) * 2;
        uint32_t v = 0;
        if (rowOK)
          v = *(const uint32_t*)&src[(((size_t)b * nc + cb + ci) * Hh + yy) * Ww
                                     + x0 + px2];
        sT[row][px2 + 1][ci] = (unsigned short)(v & 0xFFFFu);
        sT[row][px2 + 2][ci] = (unsigned short)(v >> 16);
      }
    }
    // edge columns: xx = x0-1 (lds col 0) and x0+64 (lds col 65)
    if (tid < 192) {
      int row = tid >> 6, rem = tid & 63;
      int ci = rem >> 1, e = rem & 1;
      int xx = e ? (x0 + 64) : (x0 - 1);
      int yy = y + row - 1;
      unsigned short v = 0;
      if (yy >= 0 && yy < Hh && xx >= 0 && xx < Ww)
        v = src[(((size_t)b * nc + cb + ci) * Hh + yy) * Ww + xx];
      sT[row][e ? 65 : 0][ci] = v;
    }
    __syncthreads();

    const unsigned short* wA = wpack + (size_t)chunk * 9 * COT * 512;
#pragma unroll
    for (int kidx = 0; kidx < 9; ++kidx) {
      const int ky = kidx / 3, kx = kidx % 3;   // tap offsets (+1 folded in)
      const v16bf bf = *(const v16bf*)(bBase + ky * ROWS + kx * PXS);
#pragma unroll
      for (int a = 0; a < NACC; ++a) {
        const int cot = cotB + a;  // addressing only; acc index is constant
        const v16bf af =
            *(const v16bf*)(wA + ((size_t)(kidx * COT + cot) * 32 + lane) * 16);
        acc[a] = __builtin_amdgcn_wmma_f32_16x16x32_bf16(
            false, af, false, bf, (short)0, acc[a], false, false);
      }
    }
  }

  // Epilogue. C layout: lane n=lane&15, rows m = j + 8*(lane>>4).
  const int n = lane & 15;
  const int x = x0 + pxt * 16 + n;
  const int mb = (lane >> 4) * 8;
  const size_t pixOff = (size_t)y * Ww + x;
#pragma unroll
  for (int a = 0; a < NACC; ++a) {
    const int cot = cotB + a;
#pragma unroll
    for (int j = 0; j < 8; ++j) {
      const int co = cot * 16 + mb + j;
      float v = acc[a][j];
      if (MODE == 0) {
        if (co < 64) {
          outF[((size_t)b * 64 + co) * HW + pixOff] = sigm(v + bias0[co]);
        } else {
          const int c = co - 64;
          float r  = sigm(v + bias1[c]);
          float fv = auxFeat[((size_t)b * 64 + c) * HW + pixOff];
          outB[((size_t)b * 64 + c) * HW + pixOff] = f2bf(r * fv);
        }
      } else if (MODE == 1) {
        const size_t ix = ((size_t)b * 64 + co) * HW + pixOff;
        float q  = tanhf(v + bias0[co]);
        float z  = auxZ[ix];
        float fv = auxFeat[ix];
        float h  = (1.f - z) * fv + z * q;
        outF[ix] = h;
        outB[ix] = f2bf(h);
      } else {
        if (co < 48)
          outF[((size_t)b * 48 + co) * HW + pixOff] = v;
      }
    }
  }
}

// ---------------------------------------------------------------------------
// GroupNorm stats: one block per (b, group); 6 contiguous channels per group.
// stats[(b*8+g)*2] = mean, +1 = rsqrt(var+eps)
// ---------------------------------------------------------------------------
__global__ __launch_bounds__(256) void reduceGN(const float* __restrict__ traw,
                                                float* __restrict__ stats) {
  const int bg = blockIdx.x;  // 0..31
  const float* base = traw + (size_t)bg * 6 * HW;
  const int N = 6 * HW;
  float s = 0.f, ss = 0.f;
  for (int i = threadIdx.x; i < N; i += 256) {
    float v = base[i];
    s += v; ss += v * v;
  }
  __shared__ float rs[256], rq[256];
  rs[threadIdx.x] = s; rq[threadIdx.x] = ss;
  __syncthreads();
  for (int o = 128; o > 0; o >>= 1) {
    if (threadIdx.x < o) {
      rs[threadIdx.x] += rs[threadIdx.x + o];
      rq[threadIdx.x] += rq[threadIdx.x + o];
    }
    __syncthreads();
  }
  if (threadIdx.x == 0) {
    float m   = rs[0] / (float)N;
    float var = rq[0] / (float)N - m * m;
    stats[bg * 2]     = m;
    stats[bg * 2 + 1] = rsqrtf(var + 1e-5f);
  }
}

// GN + SiLU -> bf16 (64-channel buffer, channels 48..63 zeroed)
__global__ __launch_bounds__(256) void gnApply(
    const float* __restrict__ traw, const float* __restrict__ stats,
    const float* __restrict__ gs, const float* __restrict__ gb,
    unsigned short* __restrict__ outB) {
  size_t idx = (size_t)blockIdx.x * 256 + threadIdx.x;  // B*64*HW
  int b = (int)(idx / (64 * HW));
  int rem = (int)(idx % (64 * HW));
  int c = rem / HW, p = rem % HW;
  unsigned short o = 0;
  if (c < 48) {
    float v = traw[((size_t)b * 48 + c) * HW + p];
    int g = c / 6;
    float m = stats[(b * 8 + g) * 2], r = stats[(b * 8 + g) * 2 + 1];
    v = (v - m) * r * gs[c] + gb[c];
    v = v * sigm(v);
    o = f2bf(v);
  }
  outB[idx] = o;
}

// GN2 + SiLU + four 1x1 heads fused, per pixel.
__global__ __launch_bounds__(256) void finalHeads(
    const float* __restrict__ t2raw, const float* __restrict__ stats,
    const float* __restrict__ gs, const float* __restrict__ gb,
    const float* __restrict__ wd, const float* __restrict__ bd,
    const float* __restrict__ wx, const float* __restrict__ bx,
    const float* __restrict__ wy, const float* __restrict__ by,
    const float* __restrict__ wc, const float* __restrict__ bc,
    const float* __restrict__ dI, const float* __restrict__ sxI,
    const float* __restrict__ syI, const float* __restrict__ confI,
    float* __restrict__ out) {
  int gid = blockIdx.x * 256 + threadIdx.x;
  if (gid >= Bn * HW) return;
  int b = gid / HW, p = gid % HW;
  float m[8], r[8];
#pragma unroll
  for (int g = 0; g < 8; ++g) {
    m[g] = stats[(b * 8 + g) * 2];
    r[g] = stats[(b * 8 + g) * 2 + 1];
  }
  float ad = 0.f, ax = 0.f, ay = 0.f, ac = 0.f;
#pragma unroll
  for (int c = 0; c < 48; ++c) {  // unrolled -> m/r index constant
    float v = t2raw[((size_t)b * 48 + c) * HW + p];
    int g = c / 6;
    v = (v - m[g]) * r[g] * gs[c] + gb[c];
    v = v * sigm(v);
    ad += v * wd[c]; ax += v * wx[c]; ay += v * wy[c]; ac += v * wc[c];
  }
  float sd = ad + bd[0] + dI[gid];
  out[gid]           = (sd > 20.f) ? sd : log1pf(__expf(sd));    // softplus
  out[131072 + gid]  = sxI[gid] + 0.1f * (ax + bx[0]);
  out[262144 + gid]  = syI[gid] + 0.1f * (ay + by[0]);
  out[8781824 + gid] = sigm(ac + bc[0] + 2.f * confI[gid] - 1.f);
}

// ---------------------------------------------------------------------------
extern "C" void kernel_launch(void* const* d_in, const int* in_sizes, int n_in,
                              void* d_out, int out_size, void* d_ws, size_t ws_size,
                              hipStream_t stream) {
  const float* dI    = (const float*)d_in[0];
  const float* sxI   = (const float*)d_in[1];
  const float* syI   = (const float*)d_in[2];
  const float* confI = (const float*)d_in[3];
  const float* feat  = (const float*)d_in[4];
  const float* fL    = (const float*)d_in[5];
  const float* fR    = (const float*)d_in[6];
  const float* wz    = (const float*)d_in[7];
  const float* bz    = (const float*)d_in[8];
  const float* wr    = (const float*)d_in[9];
  const float* br    = (const float*)d_in[10];
  const float* wq    = (const float*)d_in[11];
  const float* bq    = (const float*)d_in[12];
  const float* wt1   = (const float*)d_in[13];
  const float* g1s   = (const float*)d_in[14];
  const float* g1b   = (const float*)d_in[15];
  const float* wt2   = (const float*)d_in[16];
  const float* g2s   = (const float*)d_in[17];
  const float* g2b   = (const float*)d_in[18];
  const float* whd   = (const float*)d_in[19];
  const float* bhd   = (const float*)d_in[20];
  const float* whx   = (const float*)d_in[21];
  const float* bhx   = (const float*)d_in[22];
  const float* why   = (const float*)d_in[23];
  const float* bhy   = (const float*)d_in[24];
  const float* whc   = (const float*)d_in[25];
  const float* bhc   = (const float*)d_in[26];
  float* out = (float*)d_out;

  char* ws = (char*)d_ws;
  size_t off = 0;
  auto alloc = [&](size_t bytes) -> void* {
    void* p = ws + off;
    off += (bytes + 255) & ~(size_t)255;
    return p;
  };

  const size_t sz64 = (size_t)Bn * 64 * HW * 2;  // 16 MB bf16
  unsigned short* bFeat = (unsigned short*)alloc(sz64);
  unsigned short* bCtx  = (unsigned short*)alloc((size_t)Bn * 192 * HW * 2);  // 48 MB
  unsigned short* bRF   = (unsigned short*)alloc(sz64);
  unsigned short* bH    = (unsigned short*)alloc(sz64);
  float* zbuf           = (float*)alloc((size_t)Bn * 64 * HW * 4);            // 32 MB
  float* stats1         = (float*)alloc(64 * 4);
  float* stats2         = (float*)alloc(64 * 4);
  unsigned short* pZR   = (unsigned short*)alloc((size_t)72 * 8 * 512 * 2);
  unsigned short* pQ    = (unsigned short*)alloc((size_t)72 * 4 * 512 * 2);
  unsigned short* pT1   = (unsigned short*)alloc((size_t)18 * 4 * 512 * 2);
  unsigned short* pT2   = (unsigned short*)alloc((size_t)18 * 4 * 512 * 2);
  // trunk raws alias bCtx (ctx dead after the q-gate conv); 2*25.17MB == 48MB+pad
  float* t1raw = (float*)bCtx;
  float* t2raw = (float*)((char*)bCtx + (size_t)Bn * 48 * HW * 4);
  // t1-activation buffer aliases bRF (r*feat dead after the q-gate conv)
  unsigned short* bT1a = bRF;

  dim3 blk(256);
  // Weight pre-packs (cheap, deterministic, every call)
  packW<<<(72 * 8 * 32 + 255) / 256, blk, 0, stream>>>(wz, wr, pZR, 72, 8, GRUIN, 128, 64);
  packW<<<(72 * 4 * 32 + 255) / 256, blk, 0, stream>>>(wq, wq, pQ, 72, 4, GRUIN, 64, 64);
  packW<<<(18 * 4 * 32 + 255) / 256, blk, 0, stream>>>(wt1, wt1, pT1, 18, 4, 64, 48, 48);
  packW<<<(18 * 4 * 32 + 255) / 256, blk, 0, stream>>>(wt2, wt2, pT2, 18, 4, 48, 48, 48);

  // Stage 1: warp/cost/ctx assembly
  buildCtx<<<(Bn * HW + 255) / 256, blk, 0, stream>>>(dI, sxI, syI, confI, feat,
                                                      fL, fR, bFeat, bCtx);

  dim3 cgrid(Ww / 64, Hh, Bn);
  // Stage 2: fused z+r gates (236->128)
  convWMMA<0, 8, 8><<<cgrid, blk, 0, stream>>>(bFeat, 64, bCtx, 192, pZR,
                                               bz, br, feat, nullptr, zbuf, bRF);
  // Stage 3: q gate + GRU blend -> h_new (f32 into d_out, bf16 for trunk)
  convWMMA<1, 4, 8><<<cgrid, blk, 0, stream>>>(bRF, 64, bCtx, 192, pQ,
                                               bq, bq, feat, zbuf,
                                               out + 393216, bH);
  // Stage 4: trunk1 raw (64->48, padded to 64)
  convWMMA<2, 4, 2><<<cgrid, blk, 0, stream>>>(bH, 64, bH, 64, pT1,
                                               nullptr, nullptr, nullptr, nullptr,
                                               t1raw, nullptr);
  reduceGN<<<32, blk, 0, stream>>>(t1raw, stats1);
  gnApply<<<(Bn * 64 * HW) / 256, blk, 0, stream>>>(t1raw, stats1, g1s, g1b, bT1a);
  // Stage 6: trunk2 raw (48->48, padded to 64)
  convWMMA<2, 4, 2><<<cgrid, blk, 0, stream>>>(bT1a, 64, bT1a, 64, pT2,
                                               nullptr, nullptr, nullptr, nullptr,
                                               t2raw, nullptr);
  reduceGN<<<32, blk, 0, stream>>>(t2raw, stats2);
  // Stage 7: GN2 + SiLU + heads
  finalHeads<<<(Bn * HW + 255) / 256, blk, 0, stream>>>(
      t2raw, stats2, g2s, g2b, whd, bhd, whx, bhx, why, bhy, whc, bhc,
      dI, sxI, syI, confI, out);
}
